// DynamicSelfAttention_77412490543644
// MI455X (gfx1250) — matrix operand order repaired and checked by reading, not compiled, not run on hardware
//
#include <hip/hip_runtime.h>
#include <hip/hip_bf16.h>

// ---------------- types ----------------
typedef _Float16 v16h __attribute__((ext_vector_type(16)));
typedef _Float16 h8   __attribute__((ext_vector_type(8)));
typedef _Float16 h4   __attribute__((ext_vector_type(4)));
typedef float    v8f  __attribute__((ext_vector_type(8)));

// ---------------- problem constants ----------------
constexpr int  kS   = 2048;      // sequence length
constexpr int  kD   = 5120;      // model dim
constexpr int  kNH  = 40;        // heads
constexpr int  kHD  = 128;       // head dim
constexpr int  kN3  = 3 * kD;    // 15360, qkv rows

__device__ inline v16h cat8(h8 lo, h8 hi) {
  return __builtin_shufflevector(lo, hi, 0,1,2,3,4,5,6,7,8,9,10,11,12,13,14,15);
}

__device__ inline v8f wmma16(v16h a, v16h b, v8f c) {
  // D = A(16x32 f16) * B(32x16 f16) + C(16x16 f32)
  return __builtin_amdgcn_wmma_f32_16x16x32_f16(false, a, false, b, (short)0, c, false, false);
}

// ---------------- f32 -> f16 convert (vectorized x4) ----------------
__global__ __launch_bounds__(256)
void cvt_f32_f16(const float* __restrict__ x, _Float16* __restrict__ y, long n4) {
  long i = (long)blockIdx.x * blockDim.x + threadIdx.x;
  long stride = (long)gridDim.x * blockDim.x;
  for (; i < n4; i += stride) {
    float4 v = ((const float4*)x)[i];
    h4 o;
    o[0] = (_Float16)v.x; o[1] = (_Float16)v.y;
    o[2] = (_Float16)v.z; o[3] = (_Float16)v.w;
    ((h4*)y)[i] = o;
  }
}

// ---------------- NT GEMM: C[m,n] = dot(A[m,:K], B[n,:K]) + bias[n] ----------------
// A: MxK f16 row-major, B: NxK f16 row-major, C: MxN f32.
// Block = 256 threads = 8 waves; block tile 256(M)x128(N); wave tile 64x64:
// 16 WMMA per K-step against 16 b128 fragment loads (NT layout gives A and B
// the same k order, k = 16*hi + s (this code's order, not AMD's table): one
// contiguous 16-half run per lane). Speculative global_prefetch_b8 pulls the
// next K-tile while the current one is in the tensor pipe.
__global__ __launch_bounds__(256)
void gemm_nt_f16(const _Float16* __restrict__ A, const _Float16* __restrict__ B,
                 const float* __restrict__ bias, float* __restrict__ C,
                 int M, int N, int K) {
  const int lane = threadIdx.x & 31;
  const int wave = threadIdx.x >> 5;
  const int wm   = wave & 3;         // 0..3
  const int wn   = wave >> 2;        // 0..1
  const int hi   = lane >> 4;        // 0..1
  const int l16  = lane & 15;
  const int m0   = blockIdx.y * 256 + wm * 64;
  const int n0   = blockIdx.x * 128 + wn * 64;

  v8f acc[4][4];
  #pragma unroll
  for (int f = 0; f < 4; ++f)
    #pragma unroll
    for (int g = 0; g < 4; ++g) acc[f][g] = (v8f)0.0f;

  const _Float16* arow = A + (long)(m0 + l16) * K;
  const _Float16* brow = B + (long)(n0 + l16) * K;

  for (int kt = 0; kt < K; kt += 32) {
    v16h af[4], bf[4];
    #pragma unroll
    for (int f = 0; f < 4; ++f) {
      const _Float16* ap = arow + (long)(f * 16) * K + kt;
      af[f] = cat8(*(const h8*)(ap + hi*16), *(const h8*)(ap + hi*16 + 8));
      __builtin_prefetch(ap + 32, 0, 0);   // next K-tile (speculative)
    }
    #pragma unroll
    for (int g = 0; g < 4; ++g) {
      const _Float16* bp = brow + (long)(g * 16) * K + kt + hi*16;
      bf[g] = cat8(*(const h8*)bp, *(const h8*)(bp + 8));
      __builtin_prefetch(bp + 32, 0, 0);   // next K-tile (speculative)
    }
    #pragma unroll
    for (int f = 0; f < 4; ++f)
      #pragma unroll
      for (int g = 0; g < 4; ++g)
        acc[f][g] = wmma16(af[f], bf[g], acc[f][g]);
  }

  #pragma unroll
  for (int f = 0; f < 4; ++f)
    #pragma unroll
    for (int g = 0; g < 4; ++g) {
      const int n = n0 + g*16 + l16;
      const float bv = bias[n];
      #pragma unroll
      for (int j = 0; j < 8; ++j) {
        const int m = m0 + f*16 + hi*8 + j;
        C[(long)m * N + n] = acc[f][g][j] + bv;
      }
    }
}

// ---------------- fused RMSNorm (over full DIM) + RoPE + f16 pack ----------------
// qkv: (S, 15360) f32 (bias already added). Emits q16/k16 (normed+roped) and v16.
__global__ __launch_bounds__(256)
void rmsrope_kernel(const float* __restrict__ qkv, const float* __restrict__ cosv,
                    const float* __restrict__ sinv, const float* __restrict__ gq,
                    const float* __restrict__ gk, _Float16* __restrict__ q16,
                    _Float16* __restrict__ k16, _Float16* __restrict__ v16) {
  const int s   = blockIdx.x;
  const int tid = threadIdx.x;
  const float* row = qkv + (long)s * kN3;

  float sq = 0.f, sk = 0.f;
  for (int i = tid; i < kD; i += 256) {
    float x = row[i];      sq += x * x;
    float y = row[kD + i]; sk += y * y;
  }
  #pragma unroll
  for (int m = 16; m >= 1; m >>= 1) {
    sq += __shfl_xor(sq, m, 32);
    sk += __shfl_xor(sk, m, 32);
  }
  __shared__ float redq[8], redk[8];
  if ((tid & 31) == 0) { redq[tid >> 5] = sq; redk[tid >> 5] = sk; }
  __syncthreads();
  float tq = 0.f, tk = 0.f;
  #pragma unroll
  for (int w = 0; w < 8; ++w) { tq += redq[w]; tk += redk[w]; }
  const float rq = rsqrtf(tq / (float)kD + 1e-6f);
  const float rk = rsqrtf(tk / (float)kD + 1e-6f);

  for (int i = tid; i < kD; i += 256) {
    const int r = i & (kHD - 1);
    const float c  = cosv[s * kHD + r];
    const float sn = sinv[s * kHD + r];
    const int   pr = (r < 64) ? (i + 64) : (i - 64);
    const float sg = (r < 64) ? -1.f : 1.f;

    float xq  = row[i]  * rq * gq[i];
    float xqp = row[pr] * rq * gq[pr];
    q16[(long)s * kD + i] = (_Float16)(xq * c + sg * xqp * sn);

    float xk  = row[kD + i]  * rk * gk[i];
    float xkp = row[kD + pr] * rk * gk[pr];
    k16[(long)s * kD + i] = (_Float16)(xk * c + sg * xkp * sn);

    v16[(long)s * kD + i] = (_Float16)row[2 * kD + i];
  }
}

// ---------------- flash attention per (query-block of 128, head) ----------------
// 8 waves; each wave owns 16 query rows. Streams keys in blocks of 32 with
// online softmax. S = Q*K^T and O += P*V via WMMA. K staged row-major in LDS,
// V staged transposed (VT[d][key]) so the P*V B-fragment is contiguous per lane.
__global__ __launch_bounds__(256)
void flash_attn_kernel(const _Float16* __restrict__ q, const _Float16* __restrict__ k,
                       const _Float16* __restrict__ v, _Float16* __restrict__ o) {
  const int head = blockIdx.y;
  const int s0   = blockIdx.x * 128;
  const int tid  = threadIdx.x;
  const int lane = tid & 31, wave = tid >> 5;
  const int hi   = lane >> 4, l16 = lane & 15;
  const int hd0  = head * kHD;

  __shared__ _Float16 Kl[32 * kHD];       // 8 KB: K block, row-major per key
  __shared__ _Float16 VT[kHD * 32];       // 8 KB: V block transposed [d][key]
  __shared__ _Float16 Pl[8][16 * 32];     // 8 KB: per-wave probability tile

  // Q fragments: 16 rows x 128 (4 x K=32 frags), direct b128 loads
  v16h qa[4];
  {
    const _Float16* qp = q + (long)(s0 + wave * 16 + l16) * kD + hd0;
    #pragma unroll
    for (int f = 0; f < 4; ++f)
      qa[f] = cat8(*(const h8*)(qp + f*32 + hi*16), *(const h8*)(qp + f*32 + hi*16 + 8));
  }

  float mrun[8], lrun[8];
  v8f oacc[8];
  #pragma unroll
  for (int j = 0; j < 8; ++j) { mrun[j] = -1e30f; lrun[j] = 0.f; }
  #pragma unroll
  for (int c = 0; c < 8; ++c) oacc[c] = (v8f)0.0f;

  const float scale   = 0.08838834764831845f;  // 1/sqrt(128)
  const int   key_ld  = tid >> 3;              // 0..31
  const int   dseg    = (tid & 7) * 16;        // 0..112

  for (int kb = 0; kb < kS; kb += 32) {
    __syncthreads();
    {  // cooperative stage of K (row-major) and V (transposed)
      const _Float16* kp = k + (long)(kb + key_ld) * kD + hd0 + dseg;
      *(h8*)&Kl[key_ld * kHD + dseg]     = *(const h8*)kp;
      *(h8*)&Kl[key_ld * kHD + dseg + 8] = *(const h8*)(kp + 8);
      const _Float16* vp = v + (long)(kb + key_ld) * kD + hd0 + dseg;
      #pragma unroll
      for (int i = 0; i < 16; ++i)
        VT[(dseg + i) * 32 + key_ld] = vp[i];
      // speculative prefetch of next key block (dropped past end of tensor)
      __builtin_prefetch(kp + (long)32 * kD, 0, 0);
      __builtin_prefetch(vp + (long)32 * kD, 0, 0);
    }
    __syncthreads();

    // S tile: 16 queries x 32 keys
    v8f sc[2]; sc[0] = (v8f)0.0f; sc[1] = (v8f)0.0f;
    #pragma unroll
    for (int g = 0; g < 2; ++g)
      #pragma unroll
      for (int f = 0; f < 4; ++f) {
        const _Float16* bp = &Kl[(g*16 + l16) * kHD + f*32 + hi*16];
        sc[g] = wmma16(qa[f], cat8(*(const h8*)bp, *(const h8*)(bp + 8)), sc[g]);
      }

    // online softmax: rows live across 16-lane halves; butterfly masks 1..8
    float p0[8], p1[8], corr[8];
    #pragma unroll
    for (int j = 0; j < 8; ++j) {
      float a0 = sc[0][j] * scale, a1 = sc[1][j] * scale;
      float tm = fmaxf(a0, a1);
      #pragma unroll
      for (int m = 8; m >= 1; m >>= 1) tm = fmaxf(tm, __shfl_xor(tm, m, 32));
      float mnew = fmaxf(mrun[j], tm);
      corr[j] = __expf(mrun[j] - mnew);
      mrun[j] = mnew;
      p0[j] = __expf(a0 - mnew);
      p1[j] = __expf(a1 - mnew);
      float rs = p0[j] + p1[j];
      #pragma unroll
      for (int m = 8; m >= 1; m >>= 1) rs += __shfl_xor(rs, m, 32);
      lrun[j] = lrun[j] * corr[j] + rs;
    }
    #pragma unroll
    for (int c = 0; c < 8; ++c)
      #pragma unroll
      for (int j = 0; j < 8; ++j) oacc[c][j] *= corr[j];

    // stage P through per-wave LDS: convert C-layout -> A-fragment layout
    #pragma unroll
    for (int j = 0; j < 8; ++j) {
      const int m = hi * 8 + j;
      Pl[wave][m * 32 + l16]      = (_Float16)p0[j];
      Pl[wave][m * 32 + 16 + l16] = (_Float16)p1[j];
    }
    {
      const _Float16* pp = &Pl[wave][l16 * 32];
      v16h pa = cat8(*(const h8*)(pp + hi*16), *(const h8*)(pp + hi*16 + 8));
      #pragma unroll
      for (int c = 0; c < 8; ++c) {
        const _Float16* vb = &VT[(c*16 + l16) * 32 + hi*16];
        oacc[c] = wmma16(pa, cat8(*(const h8*)vb, *(const h8*)(vb + 8)), oacc[c]);
      }
    }
  }

  // finalize: divide by softmax denominator, emit f16 for out-projection
  #pragma unroll
  for (int c = 0; c < 8; ++c)
    #pragma unroll
    for (int j = 0; j < 8; ++j) {
      const int m = wave * 16 + hi * 8 + j;
      o[(long)(s0 + m) * kD + hd0 + c*16 + l16] = (_Float16)(oacc[c][j] / lrun[j]);
    }
}

// ---------------- host-side pipeline ----------------
extern "C" void kernel_launch(void* const* d_in, const int* in_sizes, int n_in,
                              void* d_out, int out_size, void* d_ws, size_t ws_size,
                              hipStream_t stream) {
  (void)in_sizes; (void)n_in; (void)out_size; (void)ws_size;
  const float* hidden = (const float*)d_in[0];
  const float* cosv   = (const float*)d_in[1];
  const float* sinv   = (const float*)d_in[2];
  const float* w_qkv  = (const float*)d_in[3];
  const float* b_qkv  = (const float*)d_in[4];
  const float* g_q    = (const float*)d_in[5];
  const float* g_k    = (const float*)d_in[6];
  const float* w_out  = (const float*)d_in[7];
  const float* b_out  = (const float*)d_in[8];

  char* p = (char*)d_ws;
  _Float16* h16    = (_Float16*)p; p += (size_t)kS  * kD * 2;
  _Float16* wqkv16 = (_Float16*)p; p += (size_t)kN3 * kD * 2;
  _Float16* wout16 = (_Float16*)p; p += (size_t)kD  * kD * 2;
  float*    qkvf   = (float*)p;    p += (size_t)kS  * kN3 * 4;
  _Float16* q16    = (_Float16*)p; p += (size_t)kS  * kD * 2;
  _Float16* k16    = (_Float16*)p; p += (size_t)kS  * kD * 2;
  _Float16* v16    = (_Float16*)p; p += (size_t)kS  * kD * 2;
  _Float16* a16    = (_Float16*)p; p += (size_t)kS  * kD * 2;

  {
    long n4h = (long)kS  * kD  / 4;
    long n4q = (long)kN3 * kD  / 4;
    long n4o = (long)kD  * kD  / 4;
    cvt_f32_f16<<<dim3((unsigned)((n4h + 255) / 256)), 256, 0, stream>>>(hidden, h16, n4h);
    cvt_f32_f16<<<dim3((unsigned)((n4q + 255) / 256)), 256, 0, stream>>>(w_qkv, wqkv16, n4q);
    cvt_f32_f16<<<dim3((unsigned)((n4o + 255) / 256)), 256, 0, stream>>>(w_out, wout16, n4o);
  }

  // QKV projection: (2048x5120) x (15360x5120)^T + b_qkv
  gemm_nt_f16<<<dim3(kN3 / 128, kS / 256), 256, 0, stream>>>(h16, wqkv16, b_qkv, qkvf, kS, kN3, kD);

  // RMSNorm over full DIM + RoPE + pack to f16
  rmsrope_kernel<<<dim3(kS), 256, 0, stream>>>(qkvf, cosv, sinv, g_q, g_k, q16, k16, v16);

  // flash attention: one WG per (128-query block, head)
  flash_attn_kernel<<<dim3(kS / 128, kNH), 256, 0, stream>>>(q16, k16, v16, a16);

  // output projection: (2048x5120) x (5120x5120)^T + b_out -> f32 d_out
  gemm_nt_f16<<<dim3(kD / 128, kS / 256), 256, 0, stream>>>(a16, wout16, b_out, (float*)d_out, kS, kD, kD);
}
